// PPOLSTMCommAgent_21414706938228
// MI455X (gfx1250) — compile-verified, matrix operand order checked by reading, not run on hardware
//
#include <hip/hip_runtime.h>
#include <hip/hip_bf16.h>
#include <math.h>

// Problem constants (match reference)
#define TT      128
#define BB      4096
#define TBTB    (TT * BB)          // 524288
#define HID     128
#define IN_DIM  36
#define KPAD    192                // 164 (36+128) -> 6 K-chunks of 32
#define NGATE   512

typedef __attribute__((ext_vector_type(16))) _Float16 v16h;
typedef __attribute__((ext_vector_type(8)))  float    v8f;
typedef __attribute__((ext_vector_type(4)))  int      i4;

// ---------------------------------------------------------------------------
// Optional CDNA5 async global->LDS DMA path (ASYNCcnt), guarded so the file
// compiles on toolchains that lack the builtins. Builtin signature is
// (int4 AS1*, int4 AS3*, imm offset, imm cpol) per the compiler diagnostic.
// ---------------------------------------------------------------------------
#if __has_builtin(__builtin_amdgcn_global_load_async_to_lds_b128) && \
    __has_builtin(__builtin_amdgcn_s_wait_asynccnt)
#define HAS_ASYNC 1
#endif

typedef __attribute__((address_space(1))) i4* gi4p;
typedef __attribute__((address_space(3))) i4* li4p;

// Copy `halves` f16 elements (multiple of 8*256) from global to LDS.
__device__ __forceinline__ void stage_lds(const _Float16* g, _Float16* l,
                                          int halves, int tid) {
#ifdef HAS_ASYNC
  for (int i = tid; i < (halves >> 3); i += 256)
    __builtin_amdgcn_global_load_async_to_lds_b128(
        (gi4p)(g + i * 8), (li4p)(l + i * 8), 0, 0);
#else
  const int4* gs = (const int4*)g;
  int4*       ls = (int4*)l;
  for (int i = tid; i < (halves >> 3); i += 256) ls[i] = gs[i];
#endif
}

#ifdef HAS_ASYNC
#define WAIT_ASYNC(n) __builtin_amdgcn_s_wait_asynccnt(n)
#else
#define WAIT_ASYNC(n) ((void)0)
#endif

__device__ __forceinline__ v8f wmma16(v16h a, v16h b, v8f c) {
  return __builtin_amdgcn_wmma_f32_16x16x32_f16(false, a, false, b, (short)0, c,
                                                false, false);
}

// K index inside a 32-wide chunk for half h of an A/B fragment
// (ISA 7.12.2, 16-bit 16x32 layout).
__device__ __forceinline__ int kmap(int h, int hi) {
  int v = h >> 1, p = h & 1;
  return ((v & 4) ? 16 : 0) + (hi << 3) + ((v & 3) << 1) + p;
}

// Inverse map: element (m, k) -> linear index in a fragment-major buffer
// laid out as [chunk kc][lane 0..31][half 0..15].
__device__ __forceinline__ int a_pos(int m, int k) {
  int kr = k & 31;
  int p = kr & 1, t = kr >> 1;
  int hi = (t >> 2) & 1;
  int v  = ((t >> 3) & 1) * 4 + (t & 3);
  return (k >> 5) * 512 + (hi * 16 + m) * 16 + v * 2 + p;
}

__device__ __forceinline__ float sigmoidf_(float x) { return 1.0f / (1.0f + expf(-x)); }

// ============================================================================
// Prep kernels: build f16 weight images in the exact per-lane WMMA fragment
// layout (fragment-major, K zero-padded) so runtime B loads are 2x ds_load_b128.
// dst index = ((kc*NT + nt)*512 + lane*16 + h); element = W[n][k] (row-major).
// ============================================================================
__global__ __launch_bounds__(256) void prep_swz(const float* __restrict__ W,
                                                int ldw, int Kreal, int NT,
                                                int total, _Float16* __restrict__ dst) {
  int idx = blockIdx.x * 256 + threadIdx.x;
  if (idx >= total) return;
  int frag = idx >> 9, r = idx & 511;
  int lane = r >> 4, h = r & 15;
  int kc = frag / NT, nt = frag - kc * NT;
  int k = kc * 32 + kmap(h, lane >> 4);
  int n = nt * 16 + (lane & 15);
  dst[idx] = (k < Kreal) ? (_Float16)W[(size_t)n * ldw + k] : (_Float16)0.0f;
}

__global__ __launch_bounds__(256) void prep_wcat(const float* __restrict__ w_ih,
                                                 const float* __restrict__ w_hh,
                                                 _Float16* __restrict__ dst) {
  int idx = blockIdx.x * 256 + threadIdx.x;
  if (idx >= 6 * 32 * 512) return;
  int frag = idx >> 9, r = idx & 511;
  int lane = r >> 4, h = r & 15;
  int kc = frag >> 5, nt = frag & 31;
  int k = kc * 32 + kmap(h, lane >> 4);
  int n = nt * 16 + (lane & 15);
  float v = (k < IN_DIM) ? w_ih[(size_t)n * IN_DIM + k]
          : (k < IN_DIM + HID) ? w_hh[(size_t)n * HID + (k - IN_DIM)] : 0.0f;
  dst[idx] = (_Float16)v;
}

__global__ __launch_bounds__(256) void prep_head(const float* __restrict__ aw,
                                                 const float* __restrict__ msw,
                                                 const float* __restrict__ cw,
                                                 _Float16* __restrict__ dst) {
  int idx = blockIdx.x * 256 + threadIdx.x;
  if (idx >= 4 * 2 * 512) return;
  int frag = idx >> 9, r = idx & 511;
  int lane = r >> 4, h = r & 15;
  int kc = frag >> 1, nt = frag & 1;
  int k = kc * 32 + kmap(h, lane >> 4);
  int n = nt * 16 + (lane & 15);
  float v = (n < 6)  ? aw[(size_t)n * HID + k]
          : (n < 22) ? msw[(size_t)(n - 6) * HID + k]
          : (n == 22) ? cw[k] : 0.0f;
  dst[idx] = (_Float16)v;
}

// Swizzled weight image sizes (halves)
#define SZ_W1   (1 * 16 * 512)   // 8192
#define SZ_W2   (8 * 16 * 512)   // 65536
#define SZ_W3   (8 * 8 * 512)    // 32768
#define SZ_W4   (4 * 1 * 512)    // 2048
#define SZ_WCAT (6 * 32 * 512)   // 98304
#define SZ_HEAD (4 * 2 * 512)    // 4096

// ============================================================================
// Kernel 1: encoder -> feat[TB][36]. Weights async-staged to LDS in fragment
// layout; activations scattered directly into next layer's A-fragment layout.
// Per-thread async b128 counts: w1=4, w2=32, w3=16, w4=1 (total 53).
// ============================================================================
#define ENC_SMEM_HALVES (4096 + 4096 + SZ_W1 + SZ_W2 + SZ_W3 + SZ_W4)
#define ENC_SMEM_BYTES  (ENC_SMEM_HALVES * 2)

__global__ __launch_bounds__(256) void enc_kernel(
    const float* __restrict__ image, const float* __restrict__ location,
    const int* __restrict__ message,
    const _Float16* __restrict__ w1p, const _Float16* __restrict__ w2p,
    const _Float16* __restrict__ w3p, const _Float16* __restrict__ w4p,
    const float* __restrict__ vb1, const float* __restrict__ vb2,
    const float* __restrict__ vb3, const float* __restrict__ vb4,
    const float* __restrict__ emb, const float* __restrict__ mw,
    const float* __restrict__ mb,  const float* __restrict__ lw,
    const float* __restrict__ lb,  float* __restrict__ feat) {
  extern __shared__ char smem[];
  _Float16* sX = (_Float16*)smem;      // 8 chunks (A-fragment layout)
  _Float16* sY = sX + 4096;            // 8 chunks
  _Float16* w1 = sY + 4096;
  _Float16* w2 = w1 + SZ_W1;
  _Float16* w3 = w2 + SZ_W2;
  _Float16* w4 = w3 + SZ_W3;

  const int tid  = threadIdx.x;
  const int wave = tid >> 5;
  const int lane = tid & 31;
  const int l16  = lane & 15;
  const int hi   = lane >> 4;
  const int r0   = blockIdx.x * 16;

  stage_lds(w1p, w1, SZ_W1, tid);
  stage_lds(w2p, w2, SZ_W2, tid);
  stage_lds(w3p, w3, SZ_W3, tid);
  stage_lds(w4p, w4, SZ_W4, tid);

  // stage image tile directly in A-fragment layout (chunk 0 of sX)
  for (int idx = tid; idx < 512; idx += 256) {
    int lp = idx >> 4, h = idx & 15;
    int m = lp & 15;
    int k = kmap(h, lp >> 4);
    sX[idx] = (k < 9) ? (_Float16)(image[(size_t)(r0 + m) * 9 + k] * (1.0f / 255.0f))
                      : (_Float16)0.0f;
  }
  WAIT_ASYNC(49);        // w1 resident
  __syncthreads();

  // layer1: 9->256, 1 K-chunk, each wave 2 N tiles; relu scatter -> sY
#pragma unroll
  for (int j = 0; j < 2; ++j) {
    int nt = wave * 2 + j;
    float bv = vb1[nt * 16 + l16];
    v8f acc;
#pragma unroll
    for (int v = 0; v < 8; ++v) acc[v] = bv;
    v16h a = *(const v16h*)&sX[lane * 16];
    v16h b = *(const v16h*)&w1[nt * 512 + lane * 16];
    acc = wmma16(a, b, acc);
#pragma unroll
    for (int v = 0; v < 8; ++v)
      sY[a_pos(v + hi * 8, nt * 16 + l16)] = (_Float16)fmaxf(acc[v], 0.0f);
  }
  WAIT_ASYNC(17);        // w2 resident
  __syncthreads();

  // layer2: 256->256, 8 K-chunks, each wave 2 N tiles; sY -> sX
#pragma unroll
  for (int j = 0; j < 2; ++j) {
    int nt = wave * 2 + j;
    float bv = vb2[nt * 16 + l16];
    v8f acc;
#pragma unroll
    for (int v = 0; v < 8; ++v) acc[v] = bv;
#pragma unroll
    for (int kc = 0; kc < 8; ++kc) {
      v16h a = *(const v16h*)&sY[kc * 512 + lane * 16];
      v16h b = *(const v16h*)&w2[(kc * 16 + nt) * 512 + lane * 16];
      acc = wmma16(a, b, acc);
    }
#pragma unroll
    for (int v = 0; v < 8; ++v)
      sX[a_pos(v + hi * 8, nt * 16 + l16)] = (_Float16)fmaxf(acc[v], 0.0f);
  }
  WAIT_ASYNC(1);         // w3 resident
  __syncthreads();

  // layer3: 256->128, 8 K-chunks, one N tile per wave; sX -> sY
  {
    int nt = wave;
    float bv = vb3[nt * 16 + l16];
    v8f acc;
#pragma unroll
    for (int v = 0; v < 8; ++v) acc[v] = bv;
#pragma unroll
    for (int kc = 0; kc < 8; ++kc) {
      v16h a = *(const v16h*)&sX[kc * 512 + lane * 16];
      v16h b = *(const v16h*)&w3[(kc * 8 + nt) * 512 + lane * 16];
      acc = wmma16(a, b, acc);
    }
#pragma unroll
    for (int v = 0; v < 8; ++v)
      sY[a_pos(v + hi * 8, nt * 16 + l16)] = (_Float16)fmaxf(acc[v], 0.0f);
  }
  WAIT_ASYNC(0);         // w4 resident
  __syncthreads();

  // layer4: 128->16, 4 K-chunks, wave 0 only; write image features to feat
  if (wave == 0) {
    float bv = vb4[l16];
    v8f acc;
#pragma unroll
    for (int v = 0; v < 8; ++v) acc[v] = bv;
#pragma unroll
    for (int kc = 0; kc < 4; ++kc) {
      v16h a = *(const v16h*)&sY[kc * 512 + lane * 16];
      v16h b = *(const v16h*)&w4[kc * 512 + lane * 16];
      acc = wmma16(a, b, acc);
    }
#pragma unroll
    for (int v = 0; v < 8; ++v)
      feat[(size_t)(r0 + v + hi * 8) * IN_DIM + l16] = fmaxf(acc[v], 0.0f);
  }
  // location features (16x4)
  if (tid >= 64 && tid < 128) {
    int q = tid - 64;
    int m = q >> 2, j = q & 3;
    size_t row = (size_t)(r0 + m);
    float x0 = location[row * 2 + 0] * 0.2f;
    float x1 = location[row * 2 + 1] * 0.2f;
    feat[row * IN_DIM + 16 + j] = x0 * lw[j * 2 + 0] + x1 * lw[j * 2 + 1] + lb[j];
  }
  // message features (16x16)
  {
    int m = tid >> 4, j = tid & 15;
    size_t row = (size_t)(r0 + m);
    int w = message[row];
    float s = mb[j];
#pragma unroll
    for (int k = 0; k < 16; ++k) s += emb[w * 16 + k] * mw[j * 16 + k];
    feat[row * IN_DIM + 20 + j] = fmaxf(s, 0.0f);
  }
}

// ============================================================================
// Kernel 2: persistent LSTM scan. 256 blocks x 16 batch rows x 128 steps.
// 192 KB swizzled weight async-staged once into LDS; per-step inner loop is
// pure WMMA + b128 DS reads, gate nonlinearities on the TRANS pipe.
// ============================================================================
#define SM_XH    (6 * 512)               // 3072 halves (A-fragment layout)
#define SM_GATES (16 * NGATE)            // 8192 floats
#define SM_HC    (16 * HID)              // 2048 floats
#define LSTM_SMEM_BYTES ((SZ_WCAT + SM_XH) * 2 + (SM_GATES + 2 * SM_HC + NGATE) * 4)

__global__ __launch_bounds__(256) void lstm_kernel(
    const float* __restrict__ feat, const float* __restrict__ done,
    const float* __restrict__ h0,   const float* __restrict__ c0,
    const _Float16* __restrict__ wcatp,
    const float* __restrict__ b_ih, const float* __restrict__ b_hh,
    float* __restrict__ out, size_t off_hN, size_t off_cN) {
  extern __shared__ char smem[];
  _Float16* wcat  = (_Float16*)smem;                 // SZ_WCAT halves
  _Float16* xh    = wcat + SZ_WCAT;                  // SM_XH halves
  float*    gates = (float*)(xh + SM_XH);
  float*    hbuf  = gates + SM_GATES;
  float*    cbuf  = hbuf + SM_HC;
  float*    bsum  = cbuf + SM_HC;

  const int tid  = threadIdx.x;
  const int wave = tid >> 5;
  const int lane = tid & 31;
  const int l16  = lane & 15;
  const int hi   = lane >> 4;
  const int r0   = blockIdx.x * 16;

  stage_lds(wcatp, wcat, SZ_WCAT, tid);   // 48 async b128 per thread

  for (int idx = tid; idx < NGATE; idx += 256) bsum[idx] = b_ih[idx] + b_hh[idx];
  for (int idx = tid; idx < SM_HC; idx += 256) {
    int m = idx >> 7, u = idx & 127;
    hbuf[idx] = h0[(size_t)(r0 + m) * HID + u];
    cbuf[idx] = c0[(size_t)(r0 + m) * HID + u];
  }
  for (int idx = tid; idx < SM_XH; idx += 256) xh[idx] = (_Float16)0.0f;
  WAIT_ASYNC(0);
  __syncthreads();

  for (int t = 0; t < TT; ++t) {
    // done-mask h/c; scatter masked h into recurrent K-slice of xh (A layout)
    for (int idx = tid; idx < SM_HC; idx += 256) {
      int m = idx >> 7, u = idx & 127;
      float dm = 1.0f - done[(size_t)t * BB + r0 + m];
      float hv = hbuf[idx] * dm;
      float cv = cbuf[idx] * dm;
      hbuf[idx] = hv;
      cbuf[idx] = cv;
      xh[a_pos(m, IN_DIM + u)] = (_Float16)hv;
    }
    // scatter x_t slice
    for (int idx = tid; idx < 16 * IN_DIM; idx += 256) {
      int m = idx / IN_DIM, k = idx - m * IN_DIM;
      xh[a_pos(m, k)] = (_Float16)feat[((size_t)t * BB + r0 + m) * IN_DIM + k];
    }
    if (t + 1 < TT)
      __builtin_prefetch(&feat[((size_t)(t + 1) * BB + r0) * IN_DIM], 0, 1);
    __syncthreads();

    // gates = [x_t, h] @ Wcat + bias; 8 waves x 4 N tiles, 6 K chunks
    v8f acc[4];
#pragma unroll
    for (int j = 0; j < 4; ++j) {
      float bv = bsum[(wave * 4 + j) * 16 + l16];
#pragma unroll
      for (int v = 0; v < 8; ++v) acc[j][v] = bv;
    }
#pragma unroll
    for (int kc = 0; kc < 6; ++kc) {
      v16h a = *(const v16h*)&xh[kc * 512 + lane * 16];
#pragma unroll
      for (int j = 0; j < 4; ++j) {
        int nt = wave * 4 + j;
        v16h b = *(const v16h*)&wcat[(kc * 32 + nt) * 512 + lane * 16];
        acc[j] = wmma16(a, b, acc[j]);
      }
    }
#pragma unroll
    for (int j = 0; j < 4; ++j) {
      int n0 = (wave * 4 + j) * 16;
#pragma unroll
      for (int v = 0; v < 8; ++v)
        gates[(v + hi * 8) * NGATE + n0 + l16] = acc[j][v];
    }
    __syncthreads();

    // pointwise LSTM cell
    for (int idx = tid; idx < SM_HC; idx += 256) {
      int m = idx >> 7, u = idx & 127;
      float ig = sigmoidf_(gates[m * NGATE + u]);
      float fg = sigmoidf_(gates[m * NGATE + HID + u]);
      float gg = tanhf(gates[m * NGATE + 2 * HID + u]);
      float og = sigmoidf_(gates[m * NGATE + 3 * HID + u]);
      float cv = fg * cbuf[idx] + ig * gg;
      float hv = og * tanhf(cv);
      cbuf[idx] = cv;
      hbuf[idx] = hv;
      out[((size_t)t * BB + r0 + m) * HID + u] = hv;
    }
    __syncthreads();
  }

  for (int idx = tid; idx < SM_HC; idx += 256) {
    int m = idx >> 7, u = idx & 127;
    out[off_hN + (size_t)(r0 + m) * HID + u] = hbuf[idx];
    out[off_cN + (size_t)(r0 + m) * HID + u] = cbuf[idx];
  }
}

// ============================================================================
// Kernel 3: heads. hidden[TB,128] @ padded [128 x 32] head matrix (8 KB LDS).
// ============================================================================
__global__ __launch_bounds__(256) void heads_kernel(
    const float* __restrict__ hidden, const _Float16* __restrict__ headp,
    const float* __restrict__ ab, const float* __restrict__ cb,
    const float* __restrict__ msb,
    float* __restrict__ out, size_t off_act, size_t off_msg, size_t off_val) {
  __shared__ _Float16 hw[SZ_HEAD];
  __shared__ float    hb[32];

  const int tid  = threadIdx.x;
  const int wave = tid >> 5;
  const int lane = tid & 31;
  const int l16  = lane & 15;
  const int hi   = lane >> 4;

  stage_lds(headp, hw, SZ_HEAD, tid);    // 2 async b128 per thread
  if (tid < 32)
    hb[tid] = (tid < 6) ? ab[tid] : (tid < 22) ? msb[tid - 6] : (tid == 22) ? cb[0] : 0.0f;
  WAIT_ASYNC(0);
  __syncthreads();

  const size_t m0 = ((size_t)blockIdx.x * 8 + wave) * 16;
  v8f acc[2];
#pragma unroll
  for (int j = 0; j < 2; ++j) {
    float bv = hb[j * 16 + l16];
#pragma unroll
    for (int v = 0; v < 8; ++v) acc[j][v] = bv;
  }
#pragma unroll
  for (int kc = 0; kc < 4; ++kc) {
    // A fragment straight from global f32 hidden (L2-resident stream)
    v16h a;
#pragma unroll
    for (int h = 0; h < 16; ++h)
      a[h] = (_Float16)hidden[(m0 + l16) * HID + kc * 32 + kmap(h, hi)];
#pragma unroll
    for (int j = 0; j < 2; ++j) {
      v16h b = *(const v16h*)&hw[(kc * 2 + j) * 512 + lane * 16];
      acc[j] = wmma16(a, b, acc[j]);
    }
  }
#pragma unroll
  for (int j = 0; j < 2; ++j) {
    int n = j * 16 + l16;
#pragma unroll
    for (int v = 0; v < 8; ++v) {
      size_t row = m0 + v + hi * 8;
      float val = acc[j][v];
      if (n < 6)        out[off_act + row * 6 + n] = val;
      else if (n < 22)  out[off_msg + row * 16 + (n - 6)] = val;
      else if (n == 22) out[off_val + row] = val;
    }
  }
}

// ============================================================================
extern "C" void kernel_launch(void* const* d_in, const int* in_sizes, int n_in,
                              void* d_out, int out_size, void* d_ws, size_t ws_size,
                              hipStream_t stream) {
  const float* image    = (const float*)d_in[0];
  const float* location = (const float*)d_in[1];
  const int*   message  = (const int*)  d_in[2];
  const float* done     = (const float*)d_in[3];
  const float* h0       = (const float*)d_in[4];
  const float* c0       = (const float*)d_in[5];
  const float* vw1 = (const float*)d_in[6],  *vb1 = (const float*)d_in[7];
  const float* vw2 = (const float*)d_in[8],  *vb2 = (const float*)d_in[9];
  const float* vw3 = (const float*)d_in[10], *vb3 = (const float*)d_in[11];
  const float* vw4 = (const float*)d_in[12], *vb4 = (const float*)d_in[13];
  const float* emb = (const float*)d_in[14];
  const float* mw  = (const float*)d_in[15], *mb = (const float*)d_in[16];
  const float* lw  = (const float*)d_in[17], *lb = (const float*)d_in[18];
  const float* w_ih = (const float*)d_in[19], *w_hh = (const float*)d_in[20];
  const float* b_ih = (const float*)d_in[21], *b_hh = (const float*)d_in[22];
  const float* aw  = (const float*)d_in[23], *ab = (const float*)d_in[24];
  const float* cw  = (const float*)d_in[25], *cb = (const float*)d_in[26];
  const float* msw = (const float*)d_in[27], *msb = (const float*)d_in[28];

  float* out  = (float*)d_out;

  // workspace layout: feat f32, then swizzled f16 weight images
  float*    feat  = (float*)d_ws;
  _Float16* wpool = (_Float16*)((char*)d_ws + (size_t)TBTB * IN_DIM * 4);
  _Float16* w1p   = wpool;
  _Float16* w2p   = w1p + SZ_W1;
  _Float16* w3p   = w2p + SZ_W2;
  _Float16* w4p   = w3p + SZ_W3;
  _Float16* wcatp = w4p + SZ_W4;
  _Float16* headp = wcatp + SZ_WCAT;

  const size_t off_act = (size_t)TBTB * HID;
  const size_t off_msg = off_act + (size_t)TBTB * 6;
  const size_t off_val = off_msg + (size_t)TBTB * 16;
  const size_t off_hN  = off_val + (size_t)TBTB;
  const size_t off_cN  = off_hN + (size_t)BB * HID;

  // build swizzled f16 weight images
  prep_swz<<<(SZ_W1 + 255) / 256, 256, 0, stream>>>(vw1, 9, 9, 16, SZ_W1, w1p);
  prep_swz<<<(SZ_W2 + 255) / 256, 256, 0, stream>>>(vw2, 256, 256, 16, SZ_W2, w2p);
  prep_swz<<<(SZ_W3 + 255) / 256, 256, 0, stream>>>(vw3, 256, 256, 8, SZ_W3, w3p);
  prep_swz<<<(SZ_W4 + 255) / 256, 256, 0, stream>>>(vw4, 128, 128, 1, SZ_W4, w4p);
  prep_wcat<<<(SZ_WCAT + 255) / 256, 256, 0, stream>>>(w_ih, w_hh, wcatp);
  prep_head<<<(SZ_HEAD + 255) / 256, 256, 0, stream>>>(aw, msw, cw, headp);

  enc_kernel<<<TBTB / 16, 256, ENC_SMEM_BYTES, stream>>>(
      image, location, message, w1p, w2p, w3p, w4p,
      vb1, vb2, vb3, vb4, emb, mw, mb, lw, lb, feat);

  lstm_kernel<<<BB / 16, 256, LSTM_SMEM_BYTES, stream>>>(
      feat, done, h0, c0, wcatp, b_ih, b_hh, out, off_hN, off_cN);

  heads_kernel<<<TBTB / (16 * 8), 256, 0, stream>>>(
      out /* hidden */, headp, ab, cb, msb, out, off_act, off_msg, off_val);
}
